// celltypeRNN_15616501088575
// MI455X (gfx1250) — matrix operand order repaired and checked by reading, not compiled, not run on hardware
//
#include <hip/hip_runtime.h>

// celltypeRNN on MI455X (gfx1250, wave32, WMMA).
// Precompute: WmT_bf16[k*H+j] = bf16(W_hh[j,k]*mask[j,k])   (29.5 MB, lives in L2)
// Per step t (256 sequential launches):
//   h_t = relu( x_t @ W_ih^T + b_ih + b_hh + h_{t-1} @ Wm^T )
// GEMM per step: C[64,3840] = A[64,3840](bf16 h shadow) x B[3840,3840](WmT)
// via v_wmma_f32_16x16x32_bf16, K-panels of B staged through double-buffered LDS.

typedef __bf16 bf16_t;
typedef __bf16 v8bf  __attribute__((ext_vector_type(8)));
typedef __bf16 v16bf __attribute__((ext_vector_type(16)));
typedef float  v8f   __attribute__((ext_vector_type(8)));

constexpr int B_ = 64;     // batch
constexpr int T_ = 256;    // timesteps
constexpr int I_ = 18;     // input width
constexpr int H_ = 3840;   // hidden
constexpr int KCH = 32;            // K per WMMA (bf16)
constexpr int NCHUNK = H_ / KCH;   // 120 K-chunks

__device__ __forceinline__ bf16_t f2bf(float f) {
    unsigned u = __builtin_bit_cast(unsigned, f);
    u += 0x7FFFu + ((u >> 16) & 1u);           // round-to-nearest-even
    unsigned short s = (unsigned short)(u >> 16);
    return __builtin_bit_cast(bf16_t, s);
}

// WmT[k*H + j] = bf16(W_hh[j*H + k] * mask[j*H + k])
__global__ void maskmul_kernel(const float* __restrict__ W_hh,
                               const float* __restrict__ mask,
                               bf16_t* __restrict__ WmT) {
    unsigned idx = blockIdx.x * 256u + threadIdx.x;   // < H*H = 14.7M
    unsigned k = idx / (unsigned)H_;
    unsigned j = idx % (unsigned)H_;
    size_t src = (size_t)j * H_ + k;
    WmT[idx] = f2bf(W_hh[src] * mask[src]);
}

// One recurrence step. grid = H/32 workgroups of 128 threads (4 waves).
// Workgroup: all 64 batch rows x 32 hidden cols. Wave w: M-tile w, 2 N-tiles.
__global__ void __launch_bounds__(128)
step_kernel(const float* __restrict__ x,      // [B,T,I]
            const float* __restrict__ W_ih,   // [H,I]
            const float* __restrict__ b_ih,   // [H]
            const float* __restrict__ b_hh,   // [H]
            const bf16_t* __restrict__ WmT,   // [H,H] K-major
            bf16_t* __restrict__ hbf,         // [B,H] bf16 shadow of h_{t-1}/h_t
            float* __restrict__ out,          // [B,T,H]
            int t) {
    const int tid  = threadIdx.x;
    const int lane = tid & 31;
    const int wv   = tid >> 5;          // 0..3 -> M tile
    const int j0   = blockIdx.x * 32;   // N strip base

    // [buf][K=32][N=32] bf16, double buffered (2 KB each)
    __shared__ __align__(64) bf16_t Bsm[2][KCH][32];

    v8f acc0 = {};
    v8f acc1 = {};

    if (t > 0) {
        const int m  = lane & 15;
        const int b  = wv * 16 + m;
        const int kb = (lane < 16) ? 0 : 8;          // A-layout half-wave K base
        const bf16_t* hrow = hbf + (size_t)b * H_;

        // stage K-chunk 0 of the B panel: thread -> (row = tid/4, 8 elts)
        {
            const int row = tid >> 2, seg = tid & 3;
            *(uint4*)&Bsm[0][row][seg * 8] =
                *(const uint4*)&WmT[(size_t)row * H_ + j0 + seg * 8];
        }
        __syncthreads();

        for (int c = 0; c < NCHUNK; ++c) {
            const int kk = c * KCH;
            if (c + 1 < NCHUNK) {                     // prefetch next panel into LDS
                const int row = tid >> 2, seg = tid & 3;
                *(uint4*)&Bsm[(c + 1) & 1][row][seg * 8] =
                    *(const uint4*)&WmT[(size_t)(kk + KCH + row) * H_ + j0 + seg * 8];
            }
            // A fragment 16x32 bf16 (ISA layout: lane m holds K {kb..kb+7, 16+kb..16+kb+7})
            v8bf lo = *(const v8bf*)&hrow[kk + kb];
            v8bf hi = *(const v8bf*)&hrow[kk + 16 + kb];
            v16bf a = __builtin_shufflevector(lo, hi,
                        0,1,2,3,4,5,6,7,8,9,10,11,12,13,14,15);
            // B fragments 32x16: lane L = column N=L%16, K in same order as A
            v16bf bf0, bf1;
#pragma unroll
            for (int s = 0; s < 16; ++s) {
                const int kr = (s < 8) ? (kb + s) : (16 + kb + (s - 8));
                bf0[s] = Bsm[c & 1][kr][m];
                bf1[s] = Bsm[c & 1][kr][16 + m];
            }
            acc0 = __builtin_amdgcn_wmma_f32_16x16x32_bf16(
                       false, a, false, bf0, (short)0, acc0, false, false);
            acc1 = __builtin_amdgcn_wmma_f32_16x16x32_bf16(
                       false, a, false, bf1, (short)0, acc1, false, false);
            __syncthreads();
        }
    }

    // Epilogue: input projection (I=18 dot) + biases + ReLU, store f32 + bf16 shadow.
    // C/D layout: VGPR r -> M = r + 8*(lane>=16), N = lane%16.
    const int nl = lane & 15;
    const int h0 = j0 + nl;
    const int h1 = j0 + 16 + nl;
    float w0[I_], w1[I_];
#pragma unroll
    for (int i = 0; i < I_; ++i) {
        w0[i] = W_ih[(size_t)h0 * I_ + i];
        w1[i] = W_ih[(size_t)h1 * I_ + i];
    }
    const float bias0 = b_ih[h0] + b_hh[h0];
    const float bias1 = b_ih[h1] + b_hh[h1];
    const int mbase = wv * 16 + ((lane < 16) ? 0 : 8);
#pragma unroll
    for (int r = 0; r < 8; ++r) {
        const int b = mbase + r;
        const float* xr = x + ((size_t)b * T_ + t) * I_;
        float s0 = bias0, s1 = bias1;
#pragma unroll
        for (int i = 0; i < I_; ++i) {
            float xv = xr[i];
            s0 += xv * w0[i];
            s1 += xv * w1[i];
        }
        float v0 = acc0[r] + s0; v0 = v0 > 0.f ? v0 : 0.f;
        float v1 = acc1[r] + s1; v1 = v1 > 0.f ? v1 : 0.f;
        const size_t o = ((size_t)b * T_ + t) * H_;
        out[o + h0] = v0;
        out[o + h1] = v1;
        hbf[(size_t)b * H_ + h0] = f2bf(v0);
        hbf[(size_t)b * H_ + h1] = f2bf(v1);
    }
}

extern "C" void kernel_launch(void* const* d_in, const int* in_sizes, int n_in,
                              void* d_out, int out_size, void* d_ws, size_t ws_size,
                              hipStream_t stream) {
    (void)in_sizes; (void)n_in; (void)out_size; (void)ws_size;
    const float* x     = (const float*)d_in[0];   // [B,T,I]
    const float* W_ih  = (const float*)d_in[1];   // [H,I]
    const float* b_ih  = (const float*)d_in[2];   // [H]
    const float* W_hh  = (const float*)d_in[3];   // [H,H]
    const float* b_hh  = (const float*)d_in[4];   // [H]
    const float* mask  = (const float*)d_in[5];   // [H,H]
    float* out = (float*)d_out;

    bf16_t* WmT = (bf16_t*)d_ws;                  // H*H bf16 = 29.5 MB
    bf16_t* hbf = WmT + (size_t)H_ * H_;          // B*H  bf16 = 0.47 MB

    maskmul_kernel<<<dim3((H_ * H_) / 256), dim3(256), 0, stream>>>(W_hh, mask, WmT);
    for (int t = 0; t < T_; ++t) {
        step_kernel<<<dim3(H_ / 32), dim3(128), 0, stream>>>(
            x, W_ih, b_ih, b_hh, WmT, hbf, out, t);
    }
}
